// MultiHeadAttention_1855425871983
// MI455X (gfx1250) — matrix operand order repaired and checked by reading, not compiled, not run on hardware
//
#include <hip/hip_runtime.h>
#include <hip/hip_bf16.h>
#include <math.h>

// ---------------------------------------------------------------------------
// MHA for MI455X (gfx1250, wave32): WMMA 16x16x32 bf16 + Tensor Data Mover
//   B=2, S=2048, D_MODEL=1024, H=16, d_k=64
// d_out = [output fp32 (B*S*D)] ++ [attn fp32 (B*H*S*S)]
// ---------------------------------------------------------------------------

#define D_MODEL 1024
#define NUM_HEADS 16
#define D_K 64
#define BATCH 2
#define SEQ 2048
#define NROWS (BATCH * SEQ)          // 4096
#define BH (BATCH * NUM_HEADS)       // 32

typedef __attribute__((ext_vector_type(16))) __bf16 v16bf;
typedef __attribute__((ext_vector_type(8)))  __bf16 v8bf;
typedef __attribute__((ext_vector_type(8)))  float  v8f;
typedef __attribute__((ext_vector_type(4)))  unsigned int v4u;
typedef __attribute__((ext_vector_type(8)))  int v8i;
typedef __attribute__((ext_vector_type(4)))  int v4i;

__device__ __forceinline__ v16bf bf_combine(v8bf lo, v8bf hi) {
    v16bf r;
#pragma unroll
    for (int i = 0; i < 8; ++i) { r[i] = lo[i]; r[i + 8] = hi[i]; }
    return r;
}

// A-fragment (16x32 bf16): lane (l16,half); element i<8 -> K=8*half+i,
// i>=8 -> K=16+8*half+(i-8).  rowBase points at row start (contiguous K).
__device__ __forceinline__ v16bf load_fragA(const __bf16* rowBase, int half) {
    const __bf16* p = rowBase + 8 * half;
    v8bf lo = *(const v8bf*)(p);
    v8bf hi = *(const v8bf*)(p + 16);
    return bf_combine(lo, hi);
}

// B-fragment (32x16 bf16): lane n=l16, element i<8 -> K=8*half+i, i>=8 -> K=16+8*half+(i-8).
__device__ __forceinline__ v16bf load_fragB(const __bf16* rowBase, int half) {
    const __bf16* p = rowBase + 8 * half;
    v8bf lo = *(const v8bf*)(p);
    v8bf hi = *(const v8bf*)(p + 16);
    return bf_combine(lo, hi);
}

// ---------------------------------------------------------------------------
// TDM: DMA a 2-D bf16 tile (tileD1 lines x tileD0 elems, line stride stride0
// elements) from global into LDS at byte offset ldsByteOff.  pad_enable with
// pad_interval=3 (16 DW = 64B) + pad_amount=3 (4 DW = 16B) gives an 80-byte
// LDS row stride (40 bf16) -> conflict-free ds_load_b128 fragment reads.
// D# layout per CDNA5 ISA §8 (groups 2/3 zero: 2-D tile).
// ---------------------------------------------------------------------------
__device__ __forceinline__ void tdm_load_tile_bf16(
    const __bf16* gsrc, unsigned int ldsByteOff,
    unsigned int tensorD0, unsigned int tensorD1, unsigned int stride0,
    unsigned int tileD0, unsigned int tileD1)
{
    unsigned long long ga = (unsigned long long)gsrc;
    v4u g0;
    g0[0] = 1u;                                        // count=1, user mode
    g0[1] = ldsByteOff;                                // lds_addr
    g0[2] = (unsigned int)ga;                          // global_addr[31:0]
    g0[3] = (unsigned int)(ga >> 32) | 0x80000000u;    // ga[56:32] | type=2
    v8i g1;
    g1[0] = (int)((1u << 16)      // data_size = 2 bytes
                | (1u << 20)      // pad_enable
                | (3u << 22)      // pad_interval: 16 DWORDs
                | (3u << 25));    // pad_amount:   4 DWORDs
    g1[1] = (int)(tensorD0 << 16);                       // tensor_dim0 lo
    g1[2] = (int)((tensorD0 >> 16) | (tensorD1 << 16));  // dim0 hi | dim1 lo
    g1[3] = (int)((tensorD1 >> 16) | (tileD0 << 16));    // dim1 hi | tile_dim0
    g1[4] = (int)tileD1;                                 // tile_dim1 (dim2=0)
    g1[5] = (int)stride0;                                // tensor_dim0_stride lo
    g1[6] = 0;
    g1[7] = 0;
    v4i z4 = {0, 0, 0, 0};
    v8i z8 = {0, 0, 0, 0, 0, 0, 0, 0};
    __builtin_amdgcn_tensor_load_to_lds(g0, g1, z4, z4, z8, 0);
}

// ---------------------------------------------------------------------------
// fp32 -> bf16 pre-convert (one-shot; makes all GEMM operands TDM-friendly)
// ---------------------------------------------------------------------------
__global__ __launch_bounds__(256) void f32_to_bf16(
    const float* __restrict__ src, __bf16* __restrict__ dst, int n)
{
    int i = (blockIdx.x * 256 + threadIdx.x) * 4;
    if (i + 3 < n) {
        float4 v = *(const float4*)(src + i);
        dst[i + 0] = (__bf16)v.x;
        dst[i + 1] = (__bf16)v.y;
        dst[i + 2] = (__bf16)v.z;
        dst[i + 3] = (__bf16)v.w;
    }
}

// ---------------------------------------------------------------------------
// GEMM  Y = A @ W^T + bias   (A bf16 [4096x1024], W bf16 [1024x1024] row-major)
// MODE 0: store bf16 split-head [bh][s][d]   (Q, K projections)
// MODE 1: store bf16 transposed [bh][d][s]   (V projection)
// MODE 2: store fp32 row-major [row][col]    (output projection)
// WG tile 32x256, 8 waves, each wave 16 rows x 64 cols (4 accumulators).
// Tiles staged by TDM (wave 0 issues, TENSORcnt-waited), double-buffered.
// ---------------------------------------------------------------------------
#define LDS_RS 40                       // bf16 row stride (80 B, TDM-padded)
#define A_TILE_BYTES (32 * 80)          // 2560
#define B_TILE_BYTES (256 * 80)         // 20480
#define GEMM_LDS (2 * A_TILE_BYTES + 2 * B_TILE_BYTES)   // 46080

template <int MODE>
__global__ __launch_bounds__(256) void gemm_tdm(
    const __bf16* __restrict__ Abf, const __bf16* __restrict__ Wbf,
    const float* __restrict__ bias, void* __restrict__ Out)
{
    extern __shared__ char gsmem[];     // dynamic LDS (starts at static size)
    const unsigned int ldsBase = (unsigned int)__builtin_amdgcn_groupstaticsize();

    const int tid  = threadIdx.x;
    const int wave = tid >> 5;
    const int lane = tid & 31;
    const int half = lane >> 4;
    const int l16  = lane & 15;
    const int rowWG = blockIdx.y * 32;
    const int colWG = blockIdx.x * 256;
    const int rblk = wave >> 2;   // 0..1
    const int cblk = wave & 3;    // 0..3

    const __bf16* Aglob = Abf + (size_t)rowWG * D_MODEL;
    const __bf16* Bglob = Wbf + (size_t)colWG * D_MODEL;

    v8f acc[4] = {};

    // prologue: chunk 0 into buffer 0
    if (wave == 0) {
        tdm_load_tile_bf16(Aglob, ldsBase,                    D_MODEL, 32,  D_MODEL, 32, 32);
        tdm_load_tile_bf16(Bglob, ldsBase + 2 * A_TILE_BYTES, D_MODEL, 256, D_MODEL, 32, 256);
    }

    for (int k0 = 0, it = 0; k0 < D_MODEL; k0 += 32, ++it) {
        const int buf = it & 1;
        if (wave == 0) {
            if (k0 + 32 < D_MODEL) {
                const int nbuf = buf ^ 1;
                const int nk = k0 + 32;
                tdm_load_tile_bf16(Aglob + nk,
                                   ldsBase + nbuf * A_TILE_BYTES,
                                   D_MODEL, 32, D_MODEL, 32, 32);
                tdm_load_tile_bf16(Bglob + nk,
                                   ldsBase + 2 * A_TILE_BYTES + nbuf * B_TILE_BYTES,
                                   D_MODEL, 256, D_MODEL, 32, 256);
                __builtin_amdgcn_s_wait_tensorcnt(2);  // chunk-k pair landed
            } else {
                __builtin_amdgcn_s_wait_tensorcnt(0);  // last chunk landed
            }
        }
        __syncthreads();   // buffer `buf` published to all waves

        const __bf16* As = (const __bf16*)(gsmem + buf * A_TILE_BYTES);
        const __bf16* Bs = (const __bf16*)(gsmem + 2 * A_TILE_BYTES + buf * B_TILE_BYTES);

        v16bf a = load_fragA(As + (rblk * 16 + l16) * LDS_RS, half);
#pragma unroll
        for (int j = 0; j < 4; ++j) {
            v16bf b = load_fragB(Bs + (cblk * 64 + j * 16 + l16) * LDS_RS, half);
            acc[j] = __builtin_amdgcn_wmma_f32_16x16x32_bf16(
                false, a, false, b, (short)0, acc[j], false, false);
        }
        __syncthreads();   // done reading `buf`; next iter may overwrite it
    }

    // ---- epilogue: D layout lane l16 -> N, vgpr r -> M = r + 8*half ----
    const int mBase = rowWG + rblk * 16 + half * 8;
#pragma unroll
    for (int j = 0; j < 4; ++j) {
        const int col = colWG + cblk * 64 + j * 16 + l16;
        const float bv = bias[col];
#pragma unroll
        for (int r = 0; r < 8; ++r) {
            float v = acc[j][r] + bv;
            const int row = mBase + r;
            if (MODE == 2) {
                ((float*)Out)[(size_t)row * D_MODEL + col] = v;
            } else {
                const int b = row >> 11, s = row & (SEQ - 1);
                const int h = col >> 6,  d = col & (D_K - 1);
                const int bh = b * NUM_HEADS + h;
                const size_t addr = (MODE == 0)
                    ? ((size_t)bh * SEQ + s) * D_K + d        // [bh][s][d]
                    : ((size_t)bh * D_K + d) * SEQ + s;       // [bh][d][s]
                ((__bf16*)Out)[addr] = (__bf16)v;
            }
        }
    }
}

// ---------------------------------------------------------------------------
// Attention: per (bh, 16-query tile) workgroup (256 thr = 8 waves).
//  Phase A: scores 16x2048 via WMMA (waves split the 128 key tiles), masked
//           and scaled, into LDS (fp32, padded stride).
//  Phase B: softmax across LDS; write attn once to global; bf16 P into LDS.
//  Phase C: context 16x64 = P @ V via WMMA (waves split the K dim),
//           cross-wave reduce with ds_add_f32, bf16 context to workspace.
// ---------------------------------------------------------------------------
#define SC_STRIDE 2052   // fp32 scores row stride (conflict-breaking pad)
#define PB_STRIDE 2056   // bf16 P row stride (16B-aligned rows)

__global__ __launch_bounds__(256) void attn_kernel(
    const __bf16* __restrict__ Q, const __bf16* __restrict__ K,
    const __bf16* __restrict__ Vt, const unsigned char* __restrict__ mask,
    float* __restrict__ attnOut, __bf16* __restrict__ ctxOut)
{
    extern __shared__ char smem[];
    float*  scores = (float*)smem;                                   // 16 x 2052 f32
    __bf16* pBf    = (__bf16*)(smem + 16 * SC_STRIDE * 4);           // 16 x 2056 bf16
    float*  ctxAcc = (float*)(smem + 16 * SC_STRIDE * 4 + 16 * PB_STRIDE * 2); // 16x64
    float*  red    = ctxAcc + 16 * 64;                               // 16x16

    const int tid  = threadIdx.x;
    const int wave = tid >> 5;
    const int lane = tid & 31;
    const int half = lane >> 4;
    const int l16  = lane & 15;
    const int qt = blockIdx.x;        // 0..127
    const int bh = blockIdx.y;        // 0..31
    const int b  = bh >> 4;
    const int qBase = qt * 16;

    for (int i = tid; i < 16 * 64; i += 256) ctxAcc[i] = 0.f;

    // ---- Phase A: scores = scale * Q K^T, masked ----
    v16bf aQ[2];
    {
        const __bf16* qrow = Q + ((size_t)bh * SEQ + qBase + l16) * D_K;
#pragma unroll
        for (int c = 0; c < 2; ++c) aQ[c] = load_fragA(qrow + c * 32, half);
    }
    const float sc = 0.125f;  // 1/sqrt(64)
    for (int kt = wave * 16; kt < wave * 16 + 16; ++kt) {
        v8f acc = {};
        const __bf16* krow = K + ((size_t)bh * SEQ + kt * 16 + l16) * D_K;
#pragma unroll
        for (int c = 0; c < 2; ++c) {
            v16bf bK = load_fragB(krow + c * 32, half);
            acc = __builtin_amdgcn_wmma_f32_16x16x32_bf16(
                false, aQ[c], false, bK, (short)0, acc, false, false);
        }
        const int kcol = kt * 16 + l16;
#pragma unroll
        for (int r = 0; r < 8; ++r) {
            const int m = r + 8 * half;
            unsigned char mv = mask[((size_t)b * SEQ + qBase + m) * SEQ + kcol];
            scores[m * SC_STRIDE + kcol] = mv ? acc[r] * sc : -1e9f;
        }
    }
    __syncthreads();

    // ---- Phase B: softmax (16 threads per row) ----
    const int row = tid >> 4;
    const int sub = tid & 15;
    float m0 = -3.4e38f;
    for (int j = sub; j < SEQ; j += 16)
        m0 = fmaxf(m0, scores[row * SC_STRIDE + j]);
    red[row * 16 + sub] = m0;
    __syncthreads();
    float rowMax = -3.4e38f;
#pragma unroll
    for (int i = 0; i < 16; ++i) rowMax = fmaxf(rowMax, red[row * 16 + i]);
    __syncthreads();
    float s0 = 0.f;
    for (int j = sub; j < SEQ; j += 16) {
        float e = __expf(scores[row * SC_STRIDE + j] - rowMax);
        scores[row * SC_STRIDE + j] = e;
        s0 += e;
    }
    red[row * 16 + sub] = s0;
    __syncthreads();
    float sum = 0.f;
#pragma unroll
    for (int i = 0; i < 16; ++i) sum += red[row * 16 + i];
    const float inv = 1.0f / sum;
    float* orow = attnOut + ((size_t)bh * SEQ + qBase + row) * SEQ;
    for (int j = sub; j < SEQ; j += 16) {
        float p = scores[row * SC_STRIDE + j] * inv;
        orow[j] = p;                        // single fp32 attn write
        pBf[row * PB_STRIDE + j] = (__bf16)p;
    }
    __syncthreads();

    // ---- Phase C: context = P @ V (waves split 64 K-chunks of 32) ----
    v8f cacc[4] = {};
    for (int c = wave * 8; c < wave * 8 + 8; ++c) {
        const int kbase = c * 32;
        v16bf aP = load_fragA(&pBf[l16 * PB_STRIDE + kbase], half);
#pragma unroll
        for (int j = 0; j < 4; ++j) {
            const __bf16* vrow = Vt + ((size_t)bh * D_K + j * 16 + l16) * SEQ + kbase;
            v16bf bV = load_fragB(vrow, half);
            cacc[j] = __builtin_amdgcn_wmma_f32_16x16x32_bf16(
                false, aP, false, bV, (short)0, cacc[j], false, false);
        }
    }
#pragma unroll
    for (int j = 0; j < 4; ++j)
#pragma unroll
        for (int r = 0; r < 8; ++r)
            atomicAdd(&ctxAcc[(r + 8 * half) * 64 + j * 16 + l16], cacc[j][r]);
    __syncthreads();

    const int h = bh & (NUM_HEADS - 1);
    for (int i = tid; i < 16 * 64; i += 256) {
        const int m = i >> 6, d = i & 63;
        ctxOut[((size_t)(b * SEQ + qBase + m)) * D_MODEL + h * D_K + d] =
            (__bf16)ctxAcc[i];
    }
}

// ---------------------------------------------------------------------------
extern "C" void kernel_launch(void* const* d_in, const int* in_sizes, int n_in,
                              void* d_out, int out_size, void* d_ws, size_t ws_size,
                              hipStream_t stream) {
    (void)in_sizes; (void)n_in; (void)out_size; (void)ws_size;
    const float* query = (const float*)d_in[0];
    const float* key_  = (const float*)d_in[1];
    const float* value = (const float*)d_in[2];
    const unsigned char* mask = (const unsigned char*)d_in[3];  // jnp.bool_ = 1B
    const float* Wq = (const float*)d_in[4];  const float* bq = (const float*)d_in[5];
    const float* Wk = (const float*)d_in[6];  const float* bk = (const float*)d_in[7];
    const float* Wv = (const float*)d_in[8];  const float* bv = (const float*)d_in[9];
    const float* Wo = (const float*)d_in[10]; const float* bo = (const float*)d_in[11];

    float* outp = (float*)d_out;                                  // [B,S,D]
    float* attn = outp + (size_t)BATCH * SEQ * D_MODEL;           // [B,H,S,S]

    const size_t xElems = (size_t)NROWS * D_MODEL;                // 4,194,304
    const size_t wElems = (size_t)D_MODEL * D_MODEL;              // 1,048,576
    __bf16* Qbf = (__bf16*)d_ws;             // [bh][s][d]
    __bf16* Kbf = Qbf + xElems;              // [bh][s][d]
    __bf16* Vt  = Kbf + xElems;              // [bh][d][s]
    __bf16* Ctx = Vt  + xElems;              // [row][col]
    __bf16* Xq  = Ctx + xElems;              // bf16 copies of inputs
    __bf16* Xk  = Xq  + xElems;
    __bf16* Xv  = Xk  + xElems;
    __bf16* Wqb = Xv  + xElems;              // bf16 copies of weights
    __bf16* Wkb = Wqb + wElems;
    __bf16* Wvb = Wkb + wElems;
    __bf16* Wob = Wvb + wElems;

    // one-shot fp32 -> bf16 conversions (TDM-friendly operands)
    const int xg = (int)(xElems / 4 / 256), wg = (int)(wElems / 4 / 256);
    f32_to_bf16<<<xg, 256, 0, stream>>>(query, Xq, (int)xElems);
    f32_to_bf16<<<xg, 256, 0, stream>>>(key_,  Xk, (int)xElems);
    f32_to_bf16<<<xg, 256, 0, stream>>>(value, Xv, (int)xElems);
    f32_to_bf16<<<wg, 256, 0, stream>>>(Wq, Wqb, (int)wElems);
    f32_to_bf16<<<wg, 256, 0, stream>>>(Wk, Wkb, (int)wElems);
    f32_to_bf16<<<wg, 256, 0, stream>>>(Wv, Wvb, (int)wElems);
    f32_to_bf16<<<wg, 256, 0, stream>>>(Wo, Wob, (int)wElems);

    dim3 gGemm(D_MODEL / 256, NROWS / 32);   // (4, 128)
    gemm_tdm<0><<<gGemm, 256, GEMM_LDS, stream>>>(Xq, Wqb, bq, Qbf);
    gemm_tdm<0><<<gGemm, 256, GEMM_LDS, stream>>>(Xk, Wkb, bk, Kbf);
    gemm_tdm<1><<<gGemm, 256, GEMM_LDS, stream>>>(Xv, Wvb, bv, Vt);

    const size_t ldsBytes =
        16 * SC_STRIDE * 4 + 16 * PB_STRIDE * 2 + 16 * 64 * 4 + 16 * 16 * 4;
    attn_kernel<<<dim3(SEQ / 16, BH), 256, ldsBytes, stream>>>(
        Qbf, Kbf, Vt, mask, attn, Ctx);

    gemm_tdm<2><<<gGemm, 256, GEMM_LDS, stream>>>(Ctx, Wob, bo, outp);
}
